// Prunus_34222299415177
// MI455X (gfx1250) — matrix operand order repaired and checked by reading, not compiled, not run on hardware
//
#include <hip/hip_runtime.h>
#include <hip/hip_bf16.h>

typedef __attribute__((ext_vector_type(16))) __bf16 v16bf;
typedef __attribute__((ext_vector_type(8)))  __bf16 v8bf;
typedef __attribute__((ext_vector_type(8)))  float  v8f;

union BF16x16 { v16bf v; v8bf h[2]; };

#define NB   65536
#define DK   3072
#define HD   128
#define TAU_INV 10.0f
#define EPSV 1e-5f

// ---------------------------------------------------------------------------
// Tile loader helpers for k_pre (fp32 global -> regs -> bf16 LDS)
// ---------------------------------------------------------------------------
__device__ __forceinline__ void loadA(const float* __restrict__ x, int rowBase,
                                      int t, int k0, float4 ra[4])
{
    const int row = t >> 1, half = (t & 1) * 16;
    const float4* src = (const float4*)(x + (size_t)(rowBase + row) * DK + k0 + half);
    #pragma unroll
    for (int i = 0; i < 4; ++i) ra[i] = src[i];
}

__device__ __forceinline__ void storeA(__bf16* As, int t, const float4 ra[4])
{
    const int row = t >> 1, half = (t & 1) * 16;
    __bf16* dst = As + row * 32 + half;
    #pragma unroll
    for (int i = 0; i < 4; ++i) {
        dst[i*4+0] = (__bf16)ra[i].x; dst[i*4+1] = (__bf16)ra[i].y;
        dst[i*4+2] = (__bf16)ra[i].z; dst[i*4+3] = (__bf16)ra[i].w;
    }
}

__device__ __forceinline__ void loadB(const float* __restrict__ W, int t, int k0,
                                      float4 rb[4])
{
    #pragma unroll
    for (int i = 0; i < 4; ++i) {
        int lin4 = i * 256 + t;                 // 1024 float4s = 32x128 tile
        int elem = lin4 * 4;
        int k = elem >> 7, col = elem & 127;
        rb[i] = *(const float4*)(W + (size_t)(k0 + k) * HD + col);
    }
}

__device__ __forceinline__ void storeB(__bf16* Bs, int t, const float4 rb[4])
{
    #pragma unroll
    for (int i = 0; i < 4; ++i) {
        int lin4 = i * 256 + t;
        int elem = lin4 * 4;
        int k = elem >> 7, col = elem & 127;
        Bs[(col+0)*32+k] = (__bf16)rb[i].x; Bs[(col+1)*32+k] = (__bf16)rb[i].y;
        Bs[(col+2)*32+k] = (__bf16)rb[i].z; Bs[(col+3)*32+k] = (__bf16)rb[i].w;
    }
}

// ---------------------------------------------------------------------------
// Kernel 1: feat = relu(LayerNorm(x @ W_pre + b_pre)), double-buffered bf16 WMMA
// Block: 128 rows x 128 cols, K = 3072 in 96 steps of 32.
// ---------------------------------------------------------------------------
__global__ __launch_bounds__(256) void k_pre(
    const float* __restrict__ x, const float* __restrict__ W_pre,
    const float* __restrict__ b_pre, const float* __restrict__ ln_g,
    const float* __restrict__ ln_b, __bf16* __restrict__ feat,
    float* __restrict__ stats)
{
    __shared__ float CtBuf[128 * 129];          // 66 KB; tiles aliased below
    __bf16* T  = (__bf16*)CtBuf;                // A0,A1 @ 0/4096; B0,B1 @ 8192/12288
    float*  Ct = CtBuf;

    const int t = threadIdx.x;
    const int rowBase = blockIdx.x * 128;
    if (blockIdx.x == 0) stats[t] = 0.0f;       // zero sum/sumsq accumulators

    const int lane = t & 31;
    const int wrow = (t >> 5) * 16;
    const int m    = lane & 15;
    const int koff = (lane & 16) ? 8  : 0;      // A-frag K offset per half-wave
    const int kb16 = (lane & 16) ? 16 : 0;      // unused; B-frag also uses koff

    v8f acc[8] = {};
    float4 ra[4], rb[4];

    // prologue: stage tile 0
    loadA(x, rowBase, t, 0, ra);
    loadB(W_pre, t, 0, rb);
    storeA(T, t, ra);
    storeB(T + 8192, t, rb);

    int cur = 0;
    for (int kt = 0; kt < 96; ++kt) {
        __syncthreads();                         // tile 'cur' visible to all
        const bool more = (kt < 95);
        if (more) {                              // issue next tile's global loads
            loadA(x, rowBase, t, (kt + 1) * 32, ra);
            loadB(W_pre, t, (kt + 1) * 32, rb);
        }
        const __bf16* As = T + cur * 4096;
        const __bf16* Bs = T + 8192 + cur * 4096;

        BF16x16 a;
        a.h[0] = *(const v8bf*)(As + (wrow + m) * 32 + koff);
        a.h[1] = *(const v8bf*)(As + (wrow + m) * 32 + koff + 16);
        #pragma unroll
        for (int c = 0; c < 8; ++c) {
            BF16x16 b;
            const int col = c * 16 + m;
            b.h[0] = *(const v8bf*)(Bs + col * 32 + koff);
            b.h[1] = *(const v8bf*)(Bs + col * 32 + koff + 16);
            acc[c] = __builtin_amdgcn_wmma_f32_16x16x32_bf16(
                false, a.v, false, b.v, (short)0, acc[c], false, false);
        }
        if (more) {                              // drain loads behind the WMMAs
            storeA(T + (cur ^ 1) * 4096, t, ra);
            storeB(T + 8192 + (cur ^ 1) * 4096, t, rb);
            cur ^= 1;
        }
    }

    __syncthreads();                             // tiles dead; reuse as f32 Ct
    const int rbase = wrow + ((lane & 16) ? 8 : 0);
    #pragma unroll
    for (int c = 0; c < 8; ++c) {
        const int col = c * 16 + m;
        #pragma unroll
        for (int r = 0; r < 8; ++r)
            Ct[(rbase + r) * 129 + col] = acc[c][r];
    }
    __syncthreads();

    if (t < 128) {                               // fused bias + LN + ReLU
        const int row = t;
        float sum = 0.f, ss = 0.f;
        for (int j = 0; j < 128; ++j) {
            float v = Ct[row * 129 + j] + b_pre[j];
            sum += v; ss += v * v;
        }
        const float mean = sum * (1.0f / 128.0f);
        const float var  = ss * (1.0f / 128.0f) - mean * mean;
        const float inv  = rsqrtf(var + EPSV);
        __bf16* out = feat + (size_t)(rowBase + row) * HD;
        for (int j = 0; j < 128; ++j) {
            float v = Ct[row * 129 + j] + b_pre[j];
            float y = (v - mean) * inv * ln_g[j] + ln_b[j];
            out[j] = (__bf16)fmaxf(y, 0.0f);
        }
    }
}

// ---------------------------------------------------------------------------
// Kernel 2: pre_d = feat @ W_disc + b_disc ; atomically accumulate col stats.
// A tile staged via CDNA5 async-to-LDS DMA (ASYNCcnt), B via fp32->bf16 LDS.
// ---------------------------------------------------------------------------
__global__ __launch_bounds__(256) void k_disc(
    const __bf16* __restrict__ feat, const float* __restrict__ W_disc,
    const float* __restrict__ b_disc, float* __restrict__ pre_d,
    float* __restrict__ stats)
{
    __shared__ float CtBuf[128 * 129];           // 66 KB, aliased
    __bf16* Fs = (__bf16*)CtBuf;                 // feat tile 128x128 bf16 (32 KB)
    __bf16* Bs = Fs + 16384;                     // W_disc^T [col][k]       (32 KB)
    float*  Ct = CtBuf;

    const int t = threadIdx.x;
    const int rowBase = blockIdx.x * 128;

    // async DMA: 32 KB feat tile, 2048 x 16B chunks, global -> LDS
    {
        const char* gbase = (const char*)(feat + (size_t)rowBase * HD);
        const unsigned lbase = (unsigned)(uintptr_t)Fs;
        #pragma unroll
        for (int i = 0; i < 8; ++i) {
            const int lin = i * 256 + t;
            const unsigned loff = lbase + lin * 16;
            const char* g = gbase + (size_t)lin * 16;
            asm volatile("global_load_async_to_lds_b128 %0, %1, off"
                         :: "v"(loff), "v"(g) : "memory");
        }
    }
    // W_disc (fp32 KxN row-major) -> bf16 LDS transposed [col][k], coalesced b128
    {
        const float4* W4 = (const float4*)W_disc;
        #pragma unroll
        for (int i = 0; i < 16; ++i) {
            int lin4 = i * 256 + t;               // 4096 float4s
            int elem = lin4 * 4;
            int k = elem >> 7, col = elem & 127;
            float4 w = W4[lin4];
            Bs[(col+0)*128+k] = (__bf16)w.x; Bs[(col+1)*128+k] = (__bf16)w.y;
            Bs[(col+2)*128+k] = (__bf16)w.z; Bs[(col+3)*128+k] = (__bf16)w.w;
        }
    }
    asm volatile("s_wait_asynccnt 0x0" ::: "memory");
    __syncthreads();

    const int lane = t & 31;
    const int wrow = (t >> 5) * 16;
    const int m    = lane & 15;
    const int koff = (lane & 16) ? 8  : 0;
    const int kb16 = (lane & 16) ? 16 : 0;

    v8f acc[8] = {};
    const __bf16* arow = Fs + (wrow + m) * 128;
    #pragma unroll
    for (int ks = 0; ks < 4; ++ks) {
        const int k0 = ks * 32;
        BF16x16 a;
        a.h[0] = *(const v8bf*)(arow + k0 + koff);
        a.h[1] = *(const v8bf*)(arow + k0 + koff + 16);
        #pragma unroll
        for (int c = 0; c < 8; ++c) {
            BF16x16 b;
            const int col = c * 16 + m;
            b.h[0] = *(const v8bf*)(Bs + col * 128 + k0 + koff);
            b.h[1] = *(const v8bf*)(Bs + col * 128 + k0 + koff + 16);
            acc[c] = __builtin_amdgcn_wmma_f32_16x16x32_bf16(
                false, a.v, false, b.v, (short)0, acc[c], false, false);
        }
    }
    __syncthreads();                              // done reading Fs/Bs alias

    const int rbase = wrow + ((lane & 16) ? 8 : 0);
    #pragma unroll
    for (int c = 0; c < 8; ++c) {
        const int col = c * 16 + m;
        const float bias = b_disc[col];
        #pragma unroll
        for (int r = 0; r < 8; ++r)
            Ct[(rbase + r) * 129 + col] = acc[c][r] + bias;
    }
    __syncthreads();

    // coalesced writeback of the 128x128 tile
    for (int i = 0; i < 64; ++i) {
        int lin = i * 256 + t;
        int row = lin >> 7, col = lin & 127;
        pre_d[(size_t)rowBase * HD + lin] = Ct[row * 129 + col];
    }
    // per-column partial sums for BatchNorm
    if (t < 128) {
        float s = 0.f, ss = 0.f;
        for (int r = 0; r < 128; ++r) {
            float v = Ct[r * 129 + t];
            s += v; ss += v * v;
        }
        atomicAdd(&stats[t], s);
        atomicAdd(&stats[128 + t], ss);
    }
}

// ---------------------------------------------------------------------------
// Kernel 3: finalize BN stats -> mean, rsqrt(var+eps)
// ---------------------------------------------------------------------------
__global__ void k_stats(float* stats)
{
    int t = threadIdx.x;
    if (t < 128) {
        float m = stats[t] * (1.0f / (float)NB);
        float v = stats[128 + t] * (1.0f / (float)NB) - m * m;
        stats[256 + t] = m;
        stats[384 + t] = rsqrtf(v + EPSV);
    }
}

// ---------------------------------------------------------------------------
// Kernel 4: BN+ReLU heads, Gumbel softmax/argmax, partitioned MLP (WMMA), gather
// ---------------------------------------------------------------------------
__global__ __launch_bounds__(256) void k_tail(
    const __bf16* __restrict__ feat, const float* __restrict__ pre_d,
    const float* __restrict__ stats, const float* __restrict__ bn_g,
    const float* __restrict__ bn_b, const float* __restrict__ W_fc,
    const float* __restrict__ b_fc, const float* __restrict__ W_sw,
    const float* __restrict__ b_sw, const float* __restrict__ gumbel,
    const float* __restrict__ Wp1, const float* __restrict__ bp1,
    const float* __restrict__ Wp2, const float* __restrict__ bp2,
    float* __restrict__ out_class, float* __restrict__ out_dom,
    float* __restrict__ out_pid, float* __restrict__ out_gum)
{
    __shared__ __bf16 Bt1[2 * 64 * 128];          // Wp1^T [p*64+j][k]  32 KB
    __shared__ __bf16 Hs[2 * 128 * 64];           // relu(feat@Wp1)     32 KB
    __shared__ int pid_s[128];

    const int t = threadIdx.x;
    const int rowBase = blockIdx.x * 128;

    // Wp1 (P,K,Hp fp32) -> bf16 LDS transposed per partition
    for (int i = 0; i < 64; ++i) {
        int lin = i * 256 + t;                    // lin = ((p*128)+k)*64 + j
        int p = lin >> 13, rem = lin & 8191;
        int k = rem >> 6, j = rem & 63;
        Bt1[(p * 64 + j) * 128 + k] = (__bf16)Wp1[lin];
    }
    __syncthreads();

    // per-row: BN+ReLU (from global pre_d) -> domain head + gumbel softmax/argmax
    if (t < 128) {
        const int grow = rowBase + t;
        const float* pr = pre_d + (size_t)grow * HD;
        float d0 = b_fc[0], d1 = b_fc[1], l0 = b_sw[0], l1 = b_sw[1];
        for (int j = 0; j < 128; ++j) {
            float v  = pr[j];
            float dv = fmaxf((v - stats[256 + j]) * stats[384 + j] * bn_g[j] + bn_b[j], 0.0f);
            d0 += dv * W_fc[2 * j];     d1 += dv * W_fc[2 * j + 1];
            l0 += dv * W_sw[2 * j];     l1 += dv * W_sw[2 * j + 1];
        }
        float g0 = gumbel[(size_t)grow * 2], g1 = gumbel[(size_t)grow * 2 + 1];
        float z0 = (l0 + g0) * TAU_INV, z1 = (l1 + g1) * TAU_INV;
        float mx = fmaxf(z0, z1);
        float e0 = __expf(z0 - mx), e1 = __expf(z1 - mx);
        float si = 1.0f / (e0 + e1);
        float y0 = e0 * si, y1 = e1 * si;
        int pid = (y1 > y0) ? 1 : 0;              // argmax, first-index on tie
        pid_s[t] = pid;
        out_dom[(size_t)grow * 2]     = d0;
        out_dom[(size_t)grow * 2 + 1] = d1;
        out_pid[grow] = (float)pid;
        float h0 = (pid == 0) ? 1.0f : 0.0f, h1 = (pid == 1) ? 1.0f : 0.0f;
        out_gum[(size_t)grow * 2]     = (h0 + y0) - y0;
        out_gum[(size_t)grow * 2 + 1] = (h1 + y1) - y1;
    }

    // WMMA: h[p] = relu(feat @ Wp1[p] + bp1[p]) for BOTH partitions (no divergence)
    const int lane = t & 31;
    const int wrow = (t >> 5) * 16;
    const int m    = lane & 15;
    const int koff = (lane & 16) ? 8  : 0;
    const int kb16 = (lane & 16) ? 16 : 0;
    const int rbase = wrow + ((lane & 16) ? 8 : 0);
    const __bf16* arow = feat + (size_t)(rowBase + wrow + m) * HD;

    for (int p = 0; p < 2; ++p) {
        v8f acc[4] = {};
        #pragma unroll
        for (int ks = 0; ks < 4; ++ks) {
            const int k0 = ks * 32;
            BF16x16 a;
            a.h[0] = *(const v8bf*)(arow + k0 + koff);
            a.h[1] = *(const v8bf*)(arow + k0 + koff + 16);
            #pragma unroll
            for (int c = 0; c < 4; ++c) {
                BF16x16 b;
                const int col = p * 64 + c * 16 + m;
                b.h[0] = *(const v8bf*)(Bt1 + col * 128 + k0 + koff);
                b.h[1] = *(const v8bf*)(Bt1 + col * 128 + k0 + koff + 16);
                acc[c] = __builtin_amdgcn_wmma_f32_16x16x32_bf16(
                    false, a.v, false, b.v, (short)0, acc[c], false, false);
            }
        }
        #pragma unroll
        for (int c = 0; c < 4; ++c) {
            const int j = c * 16 + m;
            const float bb = bp1[p * 64 + j];
            #pragma unroll
            for (int r = 0; r < 8; ++r)
                Hs[((p * 128) + rbase + r) * 64 + j] = (__bf16)fmaxf(acc[c][r] + bb, 0.0f);
        }
    }
    __syncthreads();

    // gathered tail: o = h[pid] @ Wp2[pid] + bp2[pid]   (128 rows x 10 cols)
    for (int e = t; e < 1280; e += 256) {
        int row = e / 10, cc = e - row * 10;
        int p = pid_s[row];
        float s = bp2[p * 10 + cc];
        const __bf16* hp = Hs + (p * 128 + row) * 64;
        for (int j = 0; j < 64; ++j)
            s += (float)hp[j] * Wp2[(p * 64 + j) * 10 + cc];
        out_class[(size_t)(rowBase + row) * 10 + cc] = s;
    }
}

// ---------------------------------------------------------------------------
extern "C" void kernel_launch(void* const* d_in, const int* in_sizes, int n_in,
                              void* d_out, int out_size, void* d_ws, size_t ws_size,
                              hipStream_t stream)
{
    (void)in_sizes; (void)n_in; (void)out_size; (void)ws_size;
    const float* x      = (const float*)d_in[0];
    const float* gumbel = (const float*)d_in[1];
    const float* W_pre  = (const float*)d_in[2];
    const float* b_pre  = (const float*)d_in[3];
    const float* ln_g   = (const float*)d_in[4];
    const float* ln_b   = (const float*)d_in[5];
    const float* W_disc = (const float*)d_in[6];
    const float* b_disc = (const float*)d_in[7];
    const float* bn_g   = (const float*)d_in[8];
    const float* bn_b   = (const float*)d_in[9];
    const float* W_fc   = (const float*)d_in[10];
    const float* b_fc   = (const float*)d_in[11];
    const float* W_sw   = (const float*)d_in[12];
    const float* b_sw   = (const float*)d_in[13];
    const float* Wp1    = (const float*)d_in[14];
    const float* bp1    = (const float*)d_in[15];
    const float* Wp2    = (const float*)d_in[16];
    const float* bp2    = (const float*)d_in[17];

    char* ws = (char*)d_ws;
    __bf16* feat  = (__bf16*)ws;                                      // 16 MB
    float*  pre_d = (float*)(ws + (size_t)NB * HD * sizeof(__bf16));  // 32 MB
    float*  stats = (float*)(ws + (size_t)NB * HD * (sizeof(__bf16) + sizeof(float)));

    float* out       = (float*)d_out;
    float* out_class = out;                        // B*10
    float* out_dom   = out + (size_t)NB * 10;      // B*2
    float* out_pid   = out + (size_t)NB * 12;      // B
    float* out_gum   = out + (size_t)NB * 13;      // B*2

    k_pre  <<<NB / 128, 256, 0, stream>>>(x, W_pre, b_pre, ln_g, ln_b, feat, stats);
    k_disc <<<NB / 128, 256, 0, stream>>>(feat, W_disc, b_disc, pre_d, stats);
    k_stats<<<1, 128, 0, stream>>>(stats);
    k_tail <<<NB / 128, 256, 0, stream>>>(feat, pre_d, stats, bn_g, bn_b, W_fc, b_fc,
                                          W_sw, b_sw, gumbel, Wp1, bp1, Wp2, bp2,
                                          out_class, out_dom, out_pid, out_gum);
}
